// Model_36601711297135
// MI455X (gfx1250) — matrix operand order repaired and checked by reading, not compiled, not run on hardware
//
#include <hip/hip_runtime.h>
#include <hip/hip_bf16.h>

// ---------------------------------------------------------------------------
// Problem constants (from reference): B=16, T=16, N=1024, C=48, G=32, H=150,
// A=32, P=32, OUT=1.
// ---------------------------------------------------------------------------
constexpr int BB = 16;
constexpr int TT = 16;
constexpr int NN = 1024;
constexpr int CC = 48;
constexpr int GG = 32;
constexpr int HH = 150;
constexpr int AAD = 32;   // attention dim
constexpr int H3 = 450;   // 3*H
constexpr long BN = (long)BB * NN;          // 16384 rows
constexpr float NEGF = -1e30f;

typedef __attribute__((ext_vector_type(16))) _Float16 v16h;
typedef __attribute__((ext_vector_type(8)))  float    v8f;
typedef __attribute__((ext_vector_type(4)))  int      v4i;

union F16x16 { v16h h; v4i q[2]; };

// ---------------------------------------------------------------------------
// WMMA helpers. VGPR layouts per CDNA5 ISA 7.12.2 (wave32):
//  A (16x32 f16): lane<16 holds row=lane, K in {0..7,16..23}; lane>=16 holds
//                 row=lane-16, K in {8..15,24..31}.
//  B (32x16 f16): lane<16 holds col=lane, K in {0..7,16..23}; lane>=16 col=lane-16,
//                 K in {8..15,24..31}.
//  C/D (16x16 f32): VGPR r -> (row = r + 8*(lane>=16), col = lane&15).
// ---------------------------------------------------------------------------
__device__ inline v8f wmma_f16(v16h a, v16h b, v8f c) {
  return __builtin_amdgcn_wmma_f32_16x16x32_f16(false, a, false, b,
                                                (short)0, c, false, false);
}

// A-fragment from row-major f16 LDS tile [16][32]: two contiguous 16B runs.
__device__ inline v16h frag_a_row32(const _Float16* __restrict__ p) {
  int lane = threadIdx.x & 31;
  int row  = lane & 15;
  int kb   = (lane >> 4) << 3;   // 0 or 8
  F16x16 u;
  u.q[0] = *(const v4i*)(p + row * 32 + kb);        // k = kb .. kb+7
  u.q[1] = *(const v4i*)(p + row * 32 + kb + 16);   // k = kb+16 .. kb+23
  return u.h;
}

// B-fragment from column-major f16 LDS tile [16 col][32 k]: two contiguous 16B runs.
__device__ inline v16h frag_b_col32(const _Float16* __restrict__ p) {
  int lane = threadIdx.x & 31;
  int col  = lane & 15;
  int kb   = (lane >> 4) << 3;   // 0 or 8
  F16x16 u;
  u.q[0] = *(const v4i*)(p + col * 32 + kb);
  u.q[1] = *(const v4i*)(p + col * 32 + kb + 16);
  return u.h;
}

// A-fragment from row-major f32 LDS tile [16][32] (cvt during load).
__device__ inline v16h frag_a_f32row(const float* __restrict__ p) {
  int lane = threadIdx.x & 31;
  int row  = lane & 15;
  int kb   = (lane >> 4) << 3;
  const float* r = p + row * 32;
  v16h a;
#pragma unroll
  for (int i = 0; i < 8; ++i) a[i]     = (_Float16)r[kb + i];
#pragma unroll
  for (int i = 0; i < 8; ++i) a[8 + i] = (_Float16)r[kb + 16 + i];
  return a;
}

// B-fragment from f32 LDS tile laid out [col][32 k] (cvt during load).
__device__ inline v16h frag_b_f32row(const float* __restrict__ p, int colOff) {
  int lane = threadIdx.x & 31;
  int col  = (lane & 15) + colOff;
  int kb   = (lane >> 4) << 3;
  const float* r = p + col * 32 + kb;
  v16h b;
#pragma unroll
  for (int i = 0; i < 16; ++i) b[i] = (_Float16)r[i + ((i >= 8) ? 8 : 0)];
  return b;
}

// Async global->LDS copy of one dword (CDNA5, ASYNCcnt-tracked).
__device__ inline void async_g2l_b32(void* lds, const float* gaddr) {
  unsigned loff = (unsigned)(size_t)lds;
  asm volatile("global_load_async_to_lds_b32 %0, %1, off"
               :: "v"(loff), "v"(gaddr) : "memory");
}
__device__ inline void wait_async0() {
  asm volatile("s_wait_asynccnt 0x0" ::: "memory");
}

// ---------------------------------------------------------------------------
// Generic batched WMMA GEMM:  Y[b] = act( rowScale ⊙ (A[b] @ W[b]) + bias )
//   A: rows x K row-major (lda == K).  W indexing:
//     wTrans==0 : w(k,j) = W[k*outD + j]   (K x outD)
//     wTrans==1 : w(k,j) = W[j*K + k]      (outD x K)
//   grid = (rows/16, ceil(outD/64), nbatch), block = 128 (4 waves, one
//   16-col tile per wave). K padded to multiples of 32 with zeros.
//   A tile staged f32 via async global->LDS DMA; W tile staged f16.
// ---------------------------------------------------------------------------
__global__ __launch_bounds__(128) void k_gemm(
    const float* __restrict__ A, long aStride,
    const float* __restrict__ W, long wStride,
    const float* __restrict__ bias,
    const float* __restrict__ rowScale,
    float* __restrict__ Y, long yStride,
    int rows, int K, int outD, int wTrans, int act)
{
  __shared__ __align__(16) float    As32[16 * 32];     // [row][k]
  __shared__ __align__(16) _Float16 Bs[4][16 * 32];    // [col][k] per wave

  int tid  = threadIdx.x;
  int w    = tid >> 5;
  int lane = tid & 31;
  int bz   = blockIdx.z;

  const float* Ab = A + (long)bz * aStride;
  const float* Wb = W + (long)bz * wStride;
  float*       Yb = Y + (long)bz * yStride;

  int row0    = blockIdx.x * 16;
  int colBase = (blockIdx.y * 4 + w) * 16;

  v8f acc = {0.f, 0.f, 0.f, 0.f, 0.f, 0.f, 0.f, 0.f};
  int nch = (K + 31) >> 5;

  for (int c = 0; c < nch; ++c) {
    int k0 = c << 5;
    bool fullK = (k0 + 32 <= K);
    // ---- Stage A tile (16x32 f32) via async DMA: 4 dwords/thread. --------
#pragma unroll
    for (int i = 0; i < 4; ++i) {
      int e  = tid + (i << 7);
      int r  = e >> 5, kk = e & 31;
      int kg = k0 + kk;
      const float* ga = Ab + (long)(row0 + r) * K + (fullK || kg < K ? kg : 0);
      async_g2l_b32(&As32[e], ga);
    }
    // ---- Stage this wave's W tile [col][k]: col=i, k=lane. ---------------
    if (fullK && colBase + 16 <= outD) {
      if (wTrans) {
#pragma unroll
        for (int i = 0; i < 16; ++i)
          Bs[w][i * 32 + lane] =
              (_Float16)Wb[(long)(colBase + i) * K + (k0 + lane)];
      } else {
#pragma unroll
        for (int i = 0; i < 16; ++i)
          Bs[w][i * 32 + lane] =
              (_Float16)Wb[(long)(k0 + lane) * outD + (colBase + i)];
      }
    } else {
#pragma unroll
      for (int i = 0; i < 16; ++i) {
        int kg = k0 + lane, jg = colBase + i;
        float val = 0.f;
        if (kg < K && jg < outD)
          val = wTrans ? Wb[(long)jg * K + kg] : Wb[(long)kg * outD + jg];
        Bs[w][i * 32 + lane] = (_Float16)val;
      }
    }
    wait_async0();
    if (!fullK) {   // zero-pad the A tile's out-of-range K columns
#pragma unroll
      for (int i = 0; i < 4; ++i) {
        int e = tid + (i << 7);
        if (k0 + (e & 31) >= K) As32[e] = 0.f;
      }
    }
    __syncthreads();
    v16h af = frag_a_f32row(As32);
    v16h bf = frag_b_col32(Bs[w]);
    acc = wmma_f16(af, bf, acc);
    __syncthreads();
  }

  int col  = colBase + (lane & 15);
  int roff = (lane >> 4) << 3;
  if (col < outD) {
#pragma unroll
    for (int r = 0; r < 8; ++r) {
      int   rg = row0 + r + roff;
      float vv = acc[r];
      if (rowScale) vv *= rowScale[rg];
      if (bias)     vv += bias[col];
      if (act == 1) vv = (vv > 0.f) ? vv : 0.01f * vv;   // leaky_relu(0.01)
      Yb[(long)rg * outD + col] = vv;
    }
  }
}

// ---------------------------------------------------------------------------
// X = cx * c_emb + nb * n_emb         (B,N,C)
// ---------------------------------------------------------------------------
__global__ void k_build_x(const float* __restrict__ code_x,
                          const float* __restrict__ neighbors,
                          const float* __restrict__ c_emb,
                          const float* __restrict__ n_emb,
                          float* __restrict__ X, int t)
{
  long i = (long)blockIdx.x * blockDim.x + threadIdx.x;
  if (i >= BN * CC) return;
  long bn = i / CC; int c = (int)(i % CC);
  int b = (int)(bn / NN), n = (int)(bn % NN);
  long src = ((long)b * TT + t) * NN + n;
  float cx = code_x[src], nb = neighbors[src];
  X[i] = cx * c_emb[(long)n * CC + c] + nb * n_emb[(long)n * CC + c];
}

// ---------------------------------------------------------------------------
// tc = cx*(c_emb + S), tn = nb*(n_emb + S)   where S = pr ⊙ (adj @ X)
// ---------------------------------------------------------------------------
__global__ void k_gate_inputs(const float* __restrict__ code_x,
                              const float* __restrict__ neighbors,
                              const float* __restrict__ c_emb,
                              const float* __restrict__ n_emb,
                              const float* __restrict__ S,
                              float* __restrict__ tc,
                              float* __restrict__ tn, int t)
{
  long i = (long)blockIdx.x * blockDim.x + threadIdx.x;
  if (i >= BN * CC) return;
  long bn = i / CC; int c = (int)(i % CC);
  int b = (int)(bn / NN), n = (int)(bn % NN);
  long src = ((long)b * TT + t) * NN + n;
  float cx = code_x[src], nb = neighbors[src];
  float s  = S[i];
  tc[i] = cx * (c_emb[(long)n * CC + c] + s);
  tn[i] = nb * (n_emb[(long)n * CC + c] + s);
}

// ---------------------------------------------------------------------------
// qf = (m2 ? no_prev : 0) + (m3 ? u_emb : 0)
// ---------------------------------------------------------------------------
__global__ void k_qf(const int* __restrict__ dv,
                     const float* __restrict__ no_prev,
                     const float* __restrict__ u_emb,
                     float* __restrict__ qf, int t)
{
  long i = (long)blockIdx.x * blockDim.x + threadIdx.x;
  if (i >= BN * GG) return;
  long bn = i / GG; int g = (int)(i % GG);
  int b = (int)(bn / NN), n = (int)(bn % NN);
  long dbase = (((long)b * TT + t) * NN + n) * 3;
  float v = 0.f;
  if (dv[dbase + 1] > 0) v += no_prev[i];
  if (dv[dbase + 2] > 0) v += u_emb[(long)n * GG + g];
  qf[i] = v;
}

// ---------------------------------------------------------------------------
// GRU gates: h_gru = (1-z)*tanh(i_n + r*h_n) + z*h_prev
// ---------------------------------------------------------------------------
__global__ void k_gru_gates(const float* __restrict__ gi,
                            const float* __restrict__ gh,
                            const float* __restrict__ hp,
                            float* __restrict__ hg)
{
  long i = (long)blockIdx.x * blockDim.x + threadIdx.x;
  if (i >= BN * HH) return;
  long row = i / HH; int h = (int)(i % HH);
  const float* gir = gi + row * H3;
  const float* ghr = gh + row * H3;
  float ir = gir[h], iz = gir[HH + h], in_ = gir[2 * HH + h];
  float hr = ghr[h], hz = ghr[HH + h], hn  = ghr[2 * HH + h];
  float r = 1.f / (1.f + __expf(-(ir + hr)));
  float z = 1.f / (1.f + __expf(-(iz + hz)));
  float cand = tanhf(in_ + r * hn);
  hg[i] = (1.f - z) * cand + z * hp[i];
}

// ---------------------------------------------------------------------------
// Flash attention: h_att = tanh( softmax(mask(q kᵀ / √A)) @ v )
// 4 waves/block, one 16-row m-tile each; K chunk async-DMA'd to LDS (f32),
// V chunk staged f16 transposed [h][n]; online softmax; 10 wmma acc tiles.
// grid = (N/64, B), block = 128.
// ---------------------------------------------------------------------------
__global__ __launch_bounds__(128) void k_attn(
    const float* __restrict__ q, const float* __restrict__ kmat,
    const float* __restrict__ v, const int* __restrict__ dv,
    float* __restrict__ h_att, int t)
{
  __shared__ __align__(16) float    kbuf32[32 * 32];    // [n][a]
  __shared__ __align__(16) _Float16 vbuf[160 * 32];     // [h][n] (padded h)
  __shared__ __align__(16) _Float16 pa[4][16 * 32];     // per-wave probs

  int tid = threadIdx.x, w = tid >> 5, lane = tid & 31;
  int b   = blockIdx.y;
  int m0  = (blockIdx.x * 4 + w) * 16;
  long qb = (long)b * NN * AAD;
  long vb = (long)b * NN * HH;
  const float scale = 0.17677669529663687f;   // 1/sqrt(32)

  int col = lane & 15;
  int kbA = (lane >> 4) << 3;

  // Zero the h-padding rows of vbuf once (never rewritten).
  for (int e = tid; e < 10 * 32; e += 128)
    vbuf[150 * 32 + e] = (_Float16)0.f;

  // Q fragment, loaded once (K = A = 32: exactly one chunk).
  v16h qa;
#pragma unroll
  for (int i = 0; i < 16; ++i) {
    int k = i + ((i >= 8) ? 8 : 0) + kbA;
    qa[i] = (_Float16)q[qb + (long)(m0 + col) * AAD + k];
  }

  float mrun[8], lrun[8];
  v8f acc[10];
#pragma unroll
  for (int r = 0; r < 8; ++r) { mrun[r] = NEGF; lrun[r] = 0.f; }
#pragma unroll
  for (int ct = 0; ct < 10; ++ct)
    acc[ct] = (v8f){0.f, 0.f, 0.f, 0.f, 0.f, 0.f, 0.f, 0.f};

  for (int n0 = 0; n0 < NN; n0 += 32) {
    // Stage K chunk (32x32 f32) via async DMA: 8 dwords/thread.
#pragma unroll
    for (int i = 0; i < 8; ++i) {
      int e = tid + (i << 7);
      async_g2l_b32(&kbuf32[e],
                    kmat + qb + (long)(n0 + (e >> 5)) * AAD + (e & 31));
    }
    // Stage V chunk transposed: vbuf[h][n] = v[n0+n][h].
    for (int e = tid; e < 32 * HH; e += 128) {
      int nl = e / HH, hc = e % HH;
      vbuf[hc * 32 + nl] = (_Float16)v[vb + (long)(n0 + nl) * HH + hc];
    }
    wait_async0();
    __syncthreads();

    // Scores for two 16-wide column tiles.
    v16h b0 = frag_b_f32row(kbuf32, 0);
    v16h b1 = frag_b_f32row(kbuf32, 16);
    v8f zero = {0.f, 0.f, 0.f, 0.f, 0.f, 0.f, 0.f, 0.f};
    v8f s0 = wmma_f16(qa, b0, zero);
    v8f s1 = wmma_f16(qa, b1, zero);

    int na = n0 + col, nb2 = n0 + 16 + col;
    long dbase = (((long)b * TT + t) * NN) * 3;
    int m23a = (dv[dbase + (long)na  * 3 + 1] | dv[dbase + (long)na  * 3 + 2]) > 0;
    int m23b = (dv[dbase + (long)nb2 * 3 + 1] | dv[dbase + (long)nb2 * 3 + 2]) > 0;

    int prowoff = (lane >> 4) << 3;
#pragma unroll
    for (int r = 0; r < 8; ++r) {
      float sv0 = m23a ? s0[r] * scale : NEGF;
      float sv1 = m23b ? s1[r] * scale : NEGF;
      float cm = fmaxf(sv0, sv1);
#pragma unroll
      for (int off = 8; off >= 1; off >>= 1)
        cm = fmaxf(cm, __shfl_xor(cm, off, 32));
      float newm = fmaxf(mrun[r], cm);
      float f  = __expf(mrun[r] - newm);
      float p0 = __expf(sv0 - newm);
      float p1 = __expf(sv1 - newm);
      float rs = p0 + p1;
#pragma unroll
      for (int off = 8; off >= 1; off >>= 1)
        rs += __shfl_xor(rs, off, 32);
      lrun[r] = lrun[r] * f + rs;
      mrun[r] = newm;
#pragma unroll
      for (int ct = 0; ct < 10; ++ct) acc[ct][r] *= f;
      int prow = r + prowoff;                    // D-layout row for this lane
      pa[w][prow * 32 + col]      = (_Float16)p0;
      pa[w][prow * 32 + 16 + col] = (_Float16)p1;
    }
    __syncthreads();

    // P @ V over this 32-row chunk (contiguous b128 frag loads).
    v16h pf = frag_a_row32(pa[w]);
#pragma unroll
    for (int ct = 0; ct < 10; ++ct) {
      v16h vf = frag_b_col32(vbuf + ct * 16 * 32);
      acc[ct] = wmma_f16(pf, vf, acc[ct]);
    }
    __syncthreads();
  }

  // Epilogue: normalize, tanh, store.
  int roff = (lane >> 4) << 3;
#pragma unroll
  for (int ct = 0; ct < 10; ++ct) {
    int cg = ct * 16 + col;
    if (cg < HH) {
#pragma unroll
      for (int r = 0; r < 8; ++r) {
        float vv = acc[ct][r] / lrun[r];
        h_att[vb + (long)(m0 + r + roff) * HH + cg] = tanhf(vv);
      }
    }
  }
}

// ---------------------------------------------------------------------------
// h_next = (m1?h_gru:0) + tpos*(m23?h_att:0); out_t = guarded row maxes.
// grid = B, block = 160 (col = h index).
// ---------------------------------------------------------------------------
__global__ void k_step_out(const float* __restrict__ h_gru,
                           const float* __restrict__ h_att,
                           const int* __restrict__ dv,
                           float* __restrict__ h_next,
                           float* __restrict__ outs, int t)
{
  int b = blockIdx.x;
  int col = threadIdx.x;
  if (col >= HH) return;
  float tpos = (t > 0) ? 1.f : 0.f;
  float om1 = NEGF, om23 = NEGF;
  int any1 = 0, any23 = 0;
  for (int n = 0; n < NN; ++n) {
    long dbase = (((long)b * TT + t) * NN + n) * 3;
    int m1  = dv[dbase] > 0;
    int m23 = (dv[dbase + 1] | dv[dbase + 2]) > 0;
    long idx = ((long)b * NN + n) * HH + col;
    float hg = h_gru[idx], ha = h_att[idx];
    h_next[idx] = (m1 ? hg : 0.f) + tpos * (m23 ? ha : 0.f);
    if (m1)  { om1  = fmaxf(om1,  hg); any1  = 1; }
    if (m23) { om23 = fmaxf(om23, ha); any23 = 1; }
  }
  float o = (any1 ? om1 : 0.f) + tpos * (any23 ? om23 : 0.f);
  outs[((long)b * TT + t) * HH + col] = o;
}

// ---------------------------------------------------------------------------
// Final pooling: vu = (outs@Wp + bp)@ctx; masked softmax over T; pooled@Wc+bc.
// grid = B, block = 32.
// ---------------------------------------------------------------------------
__global__ void k_final(const float* __restrict__ outs,
                        const float* __restrict__ Wp,
                        const float* __restrict__ bp,
                        const float* __restrict__ ctx,
                        const float* __restrict__ Wc,
                        const float* __restrict__ bc,
                        const int* __restrict__ lens,
                        float* __restrict__ out)
{
  int b = blockIdx.x, lane = threadIdx.x;
  int L = lens[b];
  float vu[TT];
  for (int tt = 0; tt < TT; ++tt) {
    float s = bp[lane];
    for (int h = 0; h < HH; ++h)
      s += outs[((long)b * TT + tt) * HH + h] * Wp[(long)h * 32 + lane];
    float contrib = s * ctx[lane];
#pragma unroll
    for (int off = 16; off >= 1; off >>= 1)
      contrib += __shfl_xor(contrib, off, 32);
    vu[tt] = (tt < L) ? contrib : NEGF;
  }
  float mx = NEGF;
  for (int tt = 0; tt < TT; ++tt) mx = fmaxf(mx, vu[tt]);
  float sum = 0.f, sc[TT];
  for (int tt = 0; tt < TT; ++tt) { sc[tt] = __expf(vu[tt] - mx); sum += sc[tt]; }
  float inv = 1.f / sum;
  float part = 0.f;
  for (int h = lane; h < HH; h += 32) {
    float pooled = 0.f;
    for (int tt = 0; tt < TT; ++tt)
      pooled += outs[((long)b * TT + tt) * HH + h] * sc[tt];
    part += pooled * inv * Wc[h];
  }
#pragma unroll
  for (int off = 16; off >= 1; off >>= 1)
    part += __shfl_xor(part, off, 32);
  if (lane == 0) out[b] = part + bc[0];
}

// ---------------------------------------------------------------------------
// Host orchestration.
// ---------------------------------------------------------------------------
extern "C" void kernel_launch(void* const* d_in, const int* in_sizes, int n_in,
                              void* d_out, int out_size, void* d_ws, size_t ws_size,
                              hipStream_t stream) {
  (void)in_sizes; (void)n_in; (void)out_size; (void)ws_size;
  const float* code_x    = (const float*)d_in[0];
  const int*   divided   = (const int*)  d_in[1];
  const float* neighbors = (const float*)d_in[2];
  const int*   lens      = (const int*)  d_in[3];
  const float* prior     = (const float*)d_in[4];
  const float* adj       = (const float*)d_in[5];
  const float* c_emb     = (const float*)d_in[6];
  const float* n_emb     = (const float*)d_in[7];
  const float* u_emb     = (const float*)d_in[8];
  const float* Wg        = (const float*)d_in[9];
  const float* bg        = (const float*)d_in[10];
  const float* W_ih      = (const float*)d_in[11];
  const float* b_ih      = (const float*)d_in[12];
  const float* W_hh      = (const float*)d_in[13];
  const float* b_hh      = (const float*)d_in[14];
  const float* Wq        = (const float*)d_in[15];
  const float* bq        = (const float*)d_in[16];
  const float* Wk        = (const float*)d_in[17];
  const float* bk        = (const float*)d_in[18];
  const float* Wv        = (const float*)d_in[19];
  const float* bv        = (const float*)d_in[20];
  const float* Wp        = (const float*)d_in[21];
  const float* bp        = (const float*)d_in[22];
  const float* ctx       = (const float*)d_in[23];
  const float* Wc        = (const float*)d_in[24];
  const float* bc        = (const float*)d_in[25];

  const long BNH = BN * HH, BNG = BN * GG, BNC = BN * CC, BN3 = BN * H3;

  float* ws = (float*)d_ws;
  size_t off = 0;
  auto alloc = [&](long nf) {
    float* p = ws + off;
    off += (size_t)((nf + 63) & ~63L);
    return p;
  };
  float* hA    = alloc(BNH);
  float* hB    = alloc(BNH);
  float* noA   = alloc(BNG);
  float* noB   = alloc(BNG);
  float* X     = alloc(BNC);
  float* S     = alloc(BNC);
  float* tc    = alloc(BNC);
  float* tn    = alloc(BNC);
  float* co    = alloc(BNG);
  float* qf    = alloc(BNG);
  float* qb    = alloc(BN * AAD);
  float* kb    = alloc(BN * AAD);
  float* vb    = alloc(BNH);
  float* gi    = alloc(BN3);
  float* gh    = alloc(BN3);
  float* h_gru = alloc(BNH);
  float* h_att = alloc(BNH);
  float* outs  = alloc((long)BB * TT * HH);

  hipMemsetAsync(hA,  0, BNH * sizeof(float), stream);
  hipMemsetAsync(noA, 0, BNG * sizeof(float), stream);

  float* h_prev = hA;  float* h_next = hB;
  float* no_prev = noA; float* no_next = noB;

  const int TPB = 256;
  dim3 blk(128);

  for (int t = 0; t < TT; ++t) {
    // X = cx*c_emb + nb*n_emb
    k_build_x<<<dim3((BNC + TPB - 1) / TPB), TPB, 0, stream>>>(
        code_x, neighbors, c_emb, n_emb, X, t);
    // S = pr ⊙ (adj @ X)  — batched over B, rows=N, K=N, outD=C
    k_gemm<<<dim3(NN / 16, 1, BB), blk, 0, stream>>>(
        adj, 0, X, BNC / BB, nullptr, prior, S, BNC / BB,
        NN, NN, CC, 0, 0);
    // tc / tn
    k_gate_inputs<<<dim3((BNC + TPB - 1) / TPB), TPB, 0, stream>>>(
        code_x, neighbors, c_emb, n_emb, S, tc, tn, t);
    // co = leaky(tc @ Wg + bg);  no_next = leaky(tn @ Wg + bg)
    k_gemm<<<dim3(BN / 16, 1, 1), blk, 0, stream>>>(
        tc, 0, Wg, 0, bg, nullptr, co, 0, (int)BN, CC, GG, 0, 1);
    k_gemm<<<dim3(BN / 16, 1, 1), blk, 0, stream>>>(
        tn, 0, Wg, 0, bg, nullptr, no_next, 0, (int)BN, CC, GG, 0, 1);
    // qf, q, k, v
    k_qf<<<dim3((BNG + TPB - 1) / TPB), TPB, 0, stream>>>(
        divided, no_prev, u_emb, qf, t);
    k_gemm<<<dim3(BN / 16, 1, 1), blk, 0, stream>>>(
        qf, 0, Wq, 0, bq, nullptr, qb, 0, (int)BN, GG, AAD, 0, 0);
    k_gemm<<<dim3(BN / 16, 1, 1), blk, 0, stream>>>(
        co, 0, Wk, 0, bk, nullptr, kb, 0, (int)BN, GG, AAD, 0, 0);
    k_gemm<<<dim3(BN / 16, 3, 1), blk, 0, stream>>>(
        co, 0, Wv, 0, bv, nullptr, vb, 0, (int)BN, GG, HH, 0, 0);
    // GRU projections (W stored [3H, K] -> transposed indexing)
    k_gemm<<<dim3(BN / 16, 8, 1), blk, 0, stream>>>(
        co, 0, W_ih, 0, b_ih, nullptr, gi, 0, (int)BN, GG, H3, 1, 0);
    k_gemm<<<dim3(BN / 16, 8, 1), blk, 0, stream>>>(
        h_prev, 0, W_hh, 0, b_hh, nullptr, gh, 0, (int)BN, HH, H3, 1, 0);
    k_gru_gates<<<dim3((BNH + TPB - 1) / TPB), TPB, 0, stream>>>(
        gi, gh, h_prev, h_gru);
    // Flash attention
    k_attn<<<dim3(NN / 64, BB), blk, 0, stream>>>(
        qb, kb, vb, divided, h_att, t);
    // Combine + per-step output
    k_step_out<<<dim3(BB), 160, 0, stream>>>(
        h_gru, h_att, divided, h_next, outs, t);

    float* tmp;
    tmp = h_prev;  h_prev = h_next;   h_next = tmp;
    tmp = no_prev; no_prev = no_next; no_next = tmp;
  }

  k_final<<<dim3(BB), 32, 0, stream>>>(outs, Wp, bp, ctx, Wc, bc, lens,
                                       (float*)d_out);
}
